// Attention_layer_80650895884843
// MI455X (gfx1250) — compile-verified, matrix operand order checked by reading, not run on hardware
//
#include <hip/hip_runtime.h>
#include <hip/hip_bf16.h>

#define B_DIM 256
#define S_DIM 512
#define E_DIM 512
#define U_DIM 64

typedef __attribute__((ext_vector_type(16))) __bf16 v16bf;
typedef __attribute__((ext_vector_type(2)))  __bf16 v2bf;
typedef __attribute__((ext_vector_type(8)))  float  v8f;
typedef __attribute__((ext_vector_type(4)))  unsigned int v4u;

union FragBF {
    v16bf v;
    v4u   q[2];
    unsigned int u[8];
};

__device__ __forceinline__ float fast_tanh(float x) {
#if __has_builtin(__builtin_amdgcn_tanhf)
    return __builtin_amdgcn_tanhf(x);   // v_tanh_f32 on gfx1250
#else
    return tanhf(x);
#endif
}

// fptrunc f32->bf16 (RNE) — legal type on gfx1250, lowers to hw cvt
__device__ __forceinline__ unsigned int pack_bf16x2(float a, float b) {
    v2bf r = { (__bf16)a, (__bf16)b };
    unsigned int u;
    __builtin_memcpy(&u, &r, 4);
    return u;
}

__device__ __forceinline__ unsigned short f2bf_bits(float a) {
    __bf16 h = (__bf16)a;
    unsigned short u;
    __builtin_memcpy(&u, &h, 2);
    return u;
}

__global__ __launch_bounds__(256)
void attention_fused(const float* __restrict__ dec,
                     const float* __restrict__ enc,
                     const float* __restrict__ w1,
                     const float* __restrict__ b1,
                     const float* __restrict__ w2,
                     const float* __restrict__ b2,
                     const float* __restrict__ vk,
                     const float* __restrict__ vb,
                     float* __restrict__ out_ctx,
                     float* __restrict__ out_att)
{
    __shared__ __align__(16) unsigned short w2T[U_DIM * E_DIM]; // bf16, [u][e], 64KB
    __shared__ float score[S_DIM];
    __shared__ float colAdd[U_DIM];
    __shared__ float vW[U_DIM];
    __shared__ float red[256];

    const int tid  = threadIdx.x;
    const int b    = blockIdx.x;
    const int lane = tid & 31;
    const int wave = tid >> 5;

    // ---- Stage w2^T into LDS as bf16 ----
    for (int i = tid; i < U_DIM * E_DIM; i += 256) {
        const int u = i & (U_DIM - 1);
        const int e = i >> 6;
        w2T[u * E_DIM + e] = f2bf_bits(w2[(size_t)e * U_DIM + u]);
    }
    // ---- Decoder projection: colAdd[u] = dec[b]·w1[:,u] + b1[u] + b2[u] ----
    if (tid < U_DIM) {
        float acc = b1[tid];
        const float* dr = dec + (size_t)b * 512;
        for (int e = 0; e < 512; ++e)
            acc = fmaf(dr[e], w1[(size_t)e * U_DIM + tid], acc);
        colAdd[tid] = acc + b2[tid];
        vW[tid] = vk[tid];
    }
    score[tid]       = 0.0f;
    score[tid + 256] = 0.0f;
    __syncthreads();

    // ---- WMMA: enc[b] (S x E, f32->bf16) @ w2 (E x U), fused tanh/v-dot ----
    // Each wave owns 4 exclusive m-tiles and covers ALL 4 n-tiles:
    // A loaded/packed once per K-step, reused by 4 WMMAs.
    const int   uLane = lane & 15;
    const int   kSel  = (lane >> 4) << 3;     // 0 or 8 per bf16 A-layout
    const float* encB = enc + (size_t)b * S_DIM * E_DIM;

    float cAddN[4], vUN[4];
    #pragma unroll
    for (int n = 0; n < 4; ++n) {
        cAddN[n] = colAdd[n * 16 + uLane];
        vUN[n]   = vW[n * 16 + uLane];
    }

    for (int mt = 0; mt < 4; ++mt) {
        const int m0 = (wave * 4 + mt) * 16;
        const float* aRow = encB + (size_t)(m0 + uLane) * E_DIM;
        v8f c0 = {}, c1 = {}, c2 = {}, c3 = {};
        #pragma unroll 2
        for (int k = 0; k < E_DIM; k += 32) {
            const int k0 = k + kSel;
            // A fragment: f32 global -> packed bf16 (shared by 4 WMMAs)
            const float4 a0 = *(const float4*)(aRow + k0);
            const float4 a1 = *(const float4*)(aRow + k0 + 4);
            const float4 a2 = *(const float4*)(aRow + k0 + 16);
            const float4 a3 = *(const float4*)(aRow + k0 + 20);
            FragBF A;
            A.u[0] = pack_bf16x2(a0.x, a0.y);
            A.u[1] = pack_bf16x2(a0.z, a0.w);
            A.u[2] = pack_bf16x2(a1.x, a1.y);
            A.u[3] = pack_bf16x2(a1.z, a1.w);
            A.u[4] = pack_bf16x2(a2.x, a2.y);
            A.u[5] = pack_bf16x2(a2.z, a2.w);
            A.u[6] = pack_bf16x2(a3.x, a3.y);
            A.u[7] = pack_bf16x2(a3.z, a3.w);
            // 4 B fragments (one per n-tile) from LDS, 2 x ds_load_b128 each
            FragBF B0, B1, B2, B3;
            B0.q[0] = *(const v4u*)(&w2T[(0 * 16 + uLane) * E_DIM + k0]);
            B0.q[1] = *(const v4u*)(&w2T[(0 * 16 + uLane) * E_DIM + k0 + 16]);
            B1.q[0] = *(const v4u*)(&w2T[(1 * 16 + uLane) * E_DIM + k0]);
            B1.q[1] = *(const v4u*)(&w2T[(1 * 16 + uLane) * E_DIM + k0 + 16]);
            B2.q[0] = *(const v4u*)(&w2T[(2 * 16 + uLane) * E_DIM + k0]);
            B2.q[1] = *(const v4u*)(&w2T[(2 * 16 + uLane) * E_DIM + k0 + 16]);
            B3.q[0] = *(const v4u*)(&w2T[(3 * 16 + uLane) * E_DIM + k0]);
            B3.q[1] = *(const v4u*)(&w2T[(3 * 16 + uLane) * E_DIM + k0 + 16]);
            c0 = __builtin_amdgcn_wmma_f32_16x16x32_bf16(false, A.v, false, B0.v,
                                                         (short)0, c0, false, false);
            c1 = __builtin_amdgcn_wmma_f32_16x16x32_bf16(false, A.v, false, B1.v,
                                                         (short)0, c1, false, false);
            c2 = __builtin_amdgcn_wmma_f32_16x16x32_bf16(false, A.v, false, B2.v,
                                                         (short)0, c2, false, false);
            c3 = __builtin_amdgcn_wmma_f32_16x16x32_bf16(false, A.v, false, B3.v,
                                                         (short)0, c3, false, false);
        }
        // tanh + v-weighting across the 4 n-tiles, then one LDS atomic per row
        #pragma unroll
        for (int r = 0; r < 8; ++r) {
            const int srow = m0 + r + ((lane >> 4) << 3);
            float t;
            t  = fast_tanh(c0[r] + cAddN[0]) * vUN[0];
            t += fast_tanh(c1[r] + cAddN[1]) * vUN[1];
            t += fast_tanh(c2[r] + cAddN[2]) * vUN[2];
            t += fast_tanh(c3[r] + cAddN[3]) * vUN[3];
            atomicAdd(&score[srow], t);   // 16 u-lanes combine in HW
        }
    }
    __syncthreads();

    // ---- Softmax over S ----
    const float vbias = vb[0];
    const float x0 = score[tid] + vbias;
    const float x1 = score[tid + 256] + vbias;
    red[tid] = fmaxf(x0, x1);
    __syncthreads();
    for (int off = 128; off > 0; off >>= 1) {
        if (tid < off) red[tid] = fmaxf(red[tid], red[tid + off]);
        __syncthreads();
    }
    const float mx = red[0];
    __syncthreads();
    const float e0 = __expf(x0 - mx);
    const float e1 = __expf(x1 - mx);
    red[tid] = e0 + e1;
    __syncthreads();
    for (int off = 128; off > 0; off >>= 1) {
        if (tid < off) red[tid] += red[tid + off];
        __syncthreads();
    }
    const float inv = 1.0f / red[0];
    __syncthreads();
    const float aw0 = e0 * inv;
    const float aw1 = e1 * inv;
    score[tid]       = aw0;
    score[tid + 256] = aw1;
    out_att[(size_t)b * S_DIM + tid]       = aw0;
    out_att[(size_t)b * S_DIM + tid + 256] = aw1;
    __syncthreads();

    // ---- Context vector: sum_s att[s] * enc[b,s,:] ----
    float acc0 = 0.0f, acc1 = 0.0f;
    for (int s = 0; s < S_DIM; ++s) {
        const float w = score[s];
        const float* er = encB + (size_t)s * E_DIM;
        acc0 = fmaf(w, er[tid],       acc0);
        acc1 = fmaf(w, er[tid + 256], acc1);
    }
    out_ctx[(size_t)b * E_DIM + tid]       = acc0;
    out_ctx[(size_t)b * E_DIM + tid + 256] = acc1;
}

extern "C" void kernel_launch(void* const* d_in, const int* in_sizes, int n_in,
                              void* d_out, int out_size, void* d_ws, size_t ws_size,
                              hipStream_t stream) {
    const float* dec = (const float*)d_in[0];
    const float* enc = (const float*)d_in[1];
    const float* w1  = (const float*)d_in[2];
    const float* b1  = (const float*)d_in[3];
    const float* w2  = (const float*)d_in[4];
    const float* b2  = (const float*)d_in[5];
    const float* vk  = (const float*)d_in[6];
    const float* vb  = (const float*)d_in[7];
    float* out_ctx = (float*)d_out;
    float* out_att = out_ctx + (size_t)B_DIM * E_DIM;
    attention_fused<<<B_DIM, 256, 0, stream>>>(dec, enc, w1, b1, w2, b2, vk, vb,
                                               out_ctx, out_att);
}
